// DecoderRNN_38585986188027
// MI455X (gfx1250) — compile-verified, matrix operand order checked by reading, not run on hardware
//
#include <hip/hip_runtime.h>
#include <stdint.h>

// ---------------------------------------------------------------------------
// DecoderRNN fused attention for MI455X (gfx1250, wave32, WMMA)
// B=64, S=4096, H=256, 2H=512, OUT=24
//
// Pipeline:
//   prep:   q_proj = emb_ctxt[idx] @ W_h + attn_b        (64x256)
//           emb_t  = emb_temp[idx]                       (64x256)
//           W_e -> bf16 WMMA B-fragments (256 KB)        (one-time swizzle)
//   main:   per (split,b) block: stream enc rows ONCE, project with
//           v_wmma_f32_16x16x32_bf16 (fp32 accum), v_tanh, dot v, online
//           softmax with running context accumulator  -> split partials.
//           W_e fragments staged via global_load_async_to_lds_b128.
//   comb:   merge 8 split partials per batch -> context  (64x512)
//   gru:    h = (1-sigmoid(iz+hz)) * tanh(inn + sigmoid(ir+hr)*hn)  (h0=0)
//   head:   out = h @ out_W + out_b
// ---------------------------------------------------------------------------

typedef __attribute__((ext_vector_type(16))) __bf16 v16bf;
typedef __attribute__((ext_vector_type(8)))  float  v8f;
typedef __attribute__((ext_vector_type(4)))  int    v4i;

#define GLOBAL_AS __attribute__((address_space(1)))
#define LDS_AS    __attribute__((address_space(3)))

union Frag {
    v16bf    v;
    uint32_t u[8];
    uint4    q[2];
};

#define SPLIT       8
#define ROWS_PB     (4096 / SPLIT)    /* 512 rows per block  */
#define TILES_PB    (ROWS_PB / 16)    /* 32 tiles per block  */
#define WE_U32      65536             /* 512*256 bf16 = 64K u32 = 256 KB */
#define PART_STRIDE 520               /* 512 ctx + m + l + pad */

// ---- bf16 pair packing: prefer native v_cvt_pk_bf16_f32 ----
__device__ __forceinline__ uint32_t pack_bf16(float a, float b) {
#if __has_builtin(__builtin_amdgcn_cvt_pk_bf16_f32)
    auto r = __builtin_amdgcn_cvt_pk_bf16_f32(a, b);
    return __builtin_bit_cast(uint32_t, r);
#else
    uint32_t ua = __float_as_uint(a), ub = __float_as_uint(b);
    ua += 0x7FFFu + ((ua >> 16) & 1u);   // round-to-nearest-even
    ub += 0x7FFFu + ((ub >> 16) & 1u);
    return (ua >> 16) | (ub & 0xFFFF0000u);
#endif
}

// ---- native tanh transcendental on gfx1250 (co-executes with WMMA) ----
#if __has_builtin(__builtin_amdgcn_tanhf)
#define FAST_TANH(x) __builtin_amdgcn_tanhf(x)
#else
#define FAST_TANH(x) tanhf(x)
#endif

// ---------------- prep kernels ----------------

__global__ void qproj_kernel(const int* __restrict__ fut_ctxt,
                             const float* __restrict__ emb_ctxt,
                             const float* __restrict__ attn_W,
                             const float* __restrict__ attn_b,
                             float* __restrict__ q_proj) {
    const int b = blockIdx.x, n = threadIdx.x;            // 64 x 256
    const float* e = emb_ctxt + (size_t)fut_ctxt[b] * 256;
    float s = attn_b[n];
    for (int k = 0; k < 256; ++k)                          // W_h = attn_W[0:256]
        s += e[k] * attn_W[(size_t)k * 256 + n];
    q_proj[b * 256 + n] = s;
}

__global__ void embt_kernel(const int* __restrict__ fut_temp,
                            const float* __restrict__ emb_temp,
                            float* __restrict__ emb_t) {
    const int b = blockIdx.x, h = threadIdx.x;
    emb_t[b * 256 + h] = emb_temp[(size_t)fut_temp[b] * 256 + h];
}

// W_e (rows 256..767 of attn_W) -> bf16 B-fragments.
// Fragment (cidx,n): 32 lanes x 16 bf16. Lane<16 holds N=lane,
// K = c*32 + {0..7,16..23}; lane>=16 holds N=lane-16, K = c*32 + {8..15,24..31}
// (mirror of the documented 16-bit A layout).
__global__ void wefrag_kernel(const float* __restrict__ attn_W,
                              uint32_t* __restrict__ we_frag) {
    const int frag = blockIdx.x;          // 0..255 = cidx*16 + n
    const int lane = threadIdx.x;         // 0..31
    const int cidx = frag >> 4;
    const int n    = frag & 15;
    const int hi   = lane >> 4;
    const int col  = n * 16 + (lane & 15);
    const int ka   = cidx * 32 + (hi ? 8 : 0);
    const int kb   = cidx * 32 + (hi ? 24 : 16);
    const float* W = attn_W + 256 * 256;  // W_e base
    uint32_t* dst  = we_frag + (size_t)(frag * 32 + lane) * 8;
#pragma unroll
    for (int p = 0; p < 4; ++p)
        dst[p] = pack_bf16(W[(size_t)(ka + 2 * p) * 256 + col],
                           W[(size_t)(ka + 2 * p + 1) * 256 + col]);
#pragma unroll
    for (int p = 0; p < 4; ++p)
        dst[4 + p] = pack_bf16(W[(size_t)(kb + 2 * p) * 256 + col],
                               W[(size_t)(kb + 2 * p + 1) * 256 + col]);
}

// ---------------- fused attention (the 512 MB single pass) ----------------

__global__ __launch_bounds__(256, 1)
void attn_kernel(const float* __restrict__ enc,     // (64,4096,512)
                 const float* __restrict__ vvec,    // (256)
                 const float* __restrict__ q_proj,  // (64,256)  incl. attn_b
                 const uint32_t* __restrict__ we_frag,
                 float* __restrict__ partials) {
    extern __shared__ uint32_t smem[];               // 256KB W_e + combine area
    float* comb = (float*)(smem + WE_U32);

    const int split = blockIdx.x;
    const int b     = blockIdx.y;
    const int tid   = threadIdx.x;
    const int lane  = tid & 31;
    const int wave  = tid >> 5;
    const int halfl = lane & 15;
    const int hi    = lane >> 4;

    // Stage the bf16 W_e fragments into LDS (reused 32x per block),
    // via the gfx1250 VGPR-bypassing async LDS-DMA path.
#if __has_builtin(__builtin_amdgcn_global_load_async_to_lds_b128)
    {
        GLOBAL_AS v4i* gsrc = (GLOBAL_AS v4i*)we_frag;   // addrspacecast
        LDS_AS v4i*    ldst = (LDS_AS v4i*)smem;         // addrspacecast
        for (int i = tid; i < WE_U32 / 4; i += 256)
            __builtin_amdgcn_global_load_async_to_lds_b128(gsrc + i, ldst + i, 0, 0);
#if __has_builtin(__builtin_amdgcn_s_wait_asynccnt)
        __builtin_amdgcn_s_wait_asynccnt(0);
#else
        asm volatile("s_wait_asynccnt 0x0" ::: "memory");
#endif
    }
#else
    {
        const uint4* src = (const uint4*)we_frag;
        uint4* dst = (uint4*)smem;
        for (int i = tid; i < WE_U32 / 4; i += 256) dst[i] = src[i];
    }
#endif
    __syncthreads();

    const float* encB = enc + (size_t)b * 4096 * 512;

    float m = -1e30f, l = 0.f;
    float c[16];
#pragma unroll
    for (int j = 0; j < 16; ++j) c[j] = 0.f;

    for (int t = wave; t < TILES_PB; t += 8) {
        const int rowBase = split * ROWS_PB + t * 16;
        const float* rowPtr = encB + (size_t)(rowBase + halfl) * 512;

        if (t + 8 < TILES_PB)
            __builtin_prefetch(encB + (size_t)(rowBase + 128 + halfl) * 512, 0, 0);

        v8f acc[16];
        const v8f zero = {0, 0, 0, 0, 0, 0, 0, 0};
#pragma unroll
        for (int n = 0; n < 16; ++n) acc[n] = zero;

        for (int cidx = 0; cidx < 16; ++cidx) {
            // Build the 16x32 bf16 A fragment from fp32 rows (documented layout)
            Frag a;
            const int ka = cidx * 32 + (hi ? 8 : 0);
            const int kb = cidx * 32 + (hi ? 24 : 16);
            const float4 q0 = *(const float4*)(rowPtr + ka);
            const float4 q1 = *(const float4*)(rowPtr + ka + 4);
            const float4 q2 = *(const float4*)(rowPtr + kb);
            const float4 q3 = *(const float4*)(rowPtr + kb + 4);
            a.u[0] = pack_bf16(q0.x, q0.y); a.u[1] = pack_bf16(q0.z, q0.w);
            a.u[2] = pack_bf16(q1.x, q1.y); a.u[3] = pack_bf16(q1.z, q1.w);
            a.u[4] = pack_bf16(q2.x, q2.y); a.u[5] = pack_bf16(q2.z, q2.w);
            a.u[6] = pack_bf16(q3.x, q3.y); a.u[7] = pack_bf16(q3.z, q3.w);
#pragma unroll
            for (int n = 0; n < 16; ++n) {
                Frag bf;
                const uint4* bp =
                    (const uint4*)(smem + ((size_t)(cidx * 16 + n) * 32 + lane) * 8);
                bf.q[0] = bp[0];
                bf.q[1] = bp[1];
                acc[n] = __builtin_amdgcn_wmma_f32_16x16x32_bf16(
                    false, a.v, false, bf.v, (short)0, acc[n], false, false);
            }
        }

        // energies -> per-row scores.  C layout: lane<16 holds (M=j, N=halfl+n*16),
        // lane>=16 holds (M=j+8).
        float p[8];
#pragma unroll
        for (int j = 0; j < 8; ++j) p[j] = 0.f;
#pragma unroll
        for (int n = 0; n < 16; ++n) {
            const int col = n * 16 + halfl;
            const float qv = q_proj[b * 256 + col];
            const float vv = vvec[col];
#pragma unroll
            for (int j = 0; j < 8; ++j)
                p[j] += FAST_TANH(acc[n][j] + qv) * vv;
        }
        // reduce across the 16 lanes of each half-wave (columns)
#pragma unroll
        for (int mask = 1; mask < 16; mask <<= 1)
#pragma unroll
            for (int j = 0; j < 8; ++j) p[j] += __shfl_xor(p[j], mask, 32);
        float qx[8];
#pragma unroll
        for (int j = 0; j < 8; ++j) qx[j] = __shfl_xor(p[j], 16, 32);

        float rs[16];
#pragma unroll
        for (int i = 0; i < 16; ++i)
            rs[i] = (((i >> 3) & 1) == hi) ? p[i & 7] : qx[i & 7];

        // online softmax update
        float tm = rs[0];
#pragma unroll
        for (int i = 1; i < 16; ++i) tm = fmaxf(tm, rs[i]);
        const float mn = fmaxf(m, tm);
        const float scale = __expf(m - mn);
        float w[16], sum = 0.f;
#pragma unroll
        for (int i = 0; i < 16; ++i) { w[i] = __expf(rs[i] - mn); sum += w[i]; }
        l = l * scale + sum;
        m = mn;
#pragma unroll
        for (int j = 0; j < 16; ++j) c[j] *= scale;

        // context accumulation: lane owns columns lane*16 .. lane*16+15
        const float* rp = encB + (size_t)rowBase * 512 + lane * 16;
#pragma unroll
        for (int i = 0; i < 16; ++i) {
            const float4 e0 = *(const float4*)(rp + (size_t)i * 512);
            const float4 e1 = *(const float4*)(rp + (size_t)i * 512 + 4);
            const float4 e2 = *(const float4*)(rp + (size_t)i * 512 + 8);
            const float4 e3 = *(const float4*)(rp + (size_t)i * 512 + 12);
            const float wi = w[i];
            c[0]  += wi * e0.x; c[1]  += wi * e0.y; c[2]  += wi * e0.z; c[3]  += wi * e0.w;
            c[4]  += wi * e1.x; c[5]  += wi * e1.y; c[6]  += wi * e1.z; c[7]  += wi * e1.w;
            c[8]  += wi * e2.x; c[9]  += wi * e2.y; c[10] += wi * e2.z; c[11] += wi * e2.w;
            c[12] += wi * e3.x; c[13] += wi * e3.y; c[14] += wi * e3.z; c[15] += wi * e3.w;
        }
    }

    // per-wave partials -> LDS
    float* wc = comb + wave * PART_STRIDE;
#pragma unroll
    for (int j = 0; j < 16; ++j) wc[lane * 16 + j] = c[j];
    if (lane == 0) { wc[512] = m; wc[513] = l; }
    __syncthreads();

    // block-level merge of 8 wave partials; thread owns 2 columns
    float M = -1e30f;
    for (int wv = 0; wv < 8; ++wv) M = fmaxf(M, comb[wv * PART_STRIDE + 512]);
    float L = 0.f;
    for (int wv = 0; wv < 8; ++wv)
        L += __expf(comb[wv * PART_STRIDE + 512] - M) * comb[wv * PART_STRIDE + 513];
    const int col = tid * 2;
    float c0 = 0.f, c1 = 0.f;
    for (int wv = 0; wv < 8; ++wv) {
        const float s = __expf(comb[wv * PART_STRIDE + 512] - M);
        c0 += s * comb[wv * PART_STRIDE + col];
        c1 += s * comb[wv * PART_STRIDE + col + 1];
    }
    float* part = partials + (size_t)(b * SPLIT + split) * PART_STRIDE;
    part[col] = c0;
    part[col + 1] = c1;
    if (tid == 0) { part[512] = M; part[513] = L; }
}

// ---------------- split merge -> context ----------------

__global__ void combine_kernel(const float* __restrict__ partials,
                               float* __restrict__ context) {
    const int b = blockIdx.x, tid = threadIdx.x;
    const float* pb = partials + (size_t)b * SPLIT * PART_STRIDE;
    float M = -1e30f;
    for (int p = 0; p < SPLIT; ++p) M = fmaxf(M, pb[p * PART_STRIDE + 512]);
    float L = 0.f;
    for (int p = 0; p < SPLIT; ++p)
        L += __expf(pb[p * PART_STRIDE + 512] - M) * pb[p * PART_STRIDE + 513];
    const float inv = 1.f / L;
    const int col = tid * 2;
    float c0 = 0.f, c1 = 0.f;
    for (int p = 0; p < SPLIT; ++p) {
        const float s = __expf(pb[p * PART_STRIDE + 512] - M);
        c0 += s * pb[p * PART_STRIDE + col];
        c1 += s * pb[p * PART_STRIDE + col + 1];
    }
    context[(size_t)b * 512 + col]     = c0 * inv;
    context[(size_t)b * 512 + col + 1] = c1 * inv;
}

// ---------------- GRU cell (h0 = 0) ----------------

__global__ void gru_kernel(const float* __restrict__ emb_t,
                           const float* __restrict__ context,
                           const float* __restrict__ w_ih,
                           const float* __restrict__ b_ih,
                           const float* __restrict__ b_hh,
                           float* __restrict__ d_out) {
    const int b = blockIdx.x, j = threadIdx.x;     // 64 x 256
    float ir = b_ih[j], iz = b_ih[256 + j], inn = b_ih[512 + j];
    for (int k = 0; k < 768; ++k) {
        const float x = (k < 256) ? emb_t[b * 256 + k] : context[b * 512 + (k - 256)];
        const float* wr = w_ih + (size_t)k * 768;
        ir  += x * wr[j];
        iz  += x * wr[256 + j];
        inn += x * wr[512 + j];
    }
    // h0 == 0  =>  gh = b_hh
    const float hr = b_hh[j], hz = b_hh[256 + j], hn = b_hh[512 + j];
    const float r = 1.f / (1.f + __expf(-(ir + hr)));
    const float z = 1.f / (1.f + __expf(-(iz + hz)));
    const float n = tanhf(inn + r * hn);
    const float h = (1.f - z) * n;                 // + z*h0, h0 = 0
    d_out[1536 + b * 256 + j] = h;                 // second tuple output
}

// ---------------- output head ----------------

__global__ void head_kernel(const float* __restrict__ out_W,
                            const float* __restrict__ out_b,
                            float* __restrict__ d_out) {
    const int b = blockIdx.x, o = threadIdx.x;
    if (o >= 24) return;
    const float* h = d_out + 1536 + b * 256;
    float s = out_b[o];
    for (int j = 0; j < 256; ++j) s += h[j] * out_W[j * 24 + o];
    d_out[b * 24 + o] = s;
}

// ---------------- launcher ----------------

extern "C" void kernel_launch(void* const* d_in, const int* in_sizes, int n_in,
                              void* d_out, int out_size, void* d_ws, size_t ws_size,
                              hipStream_t stream) {
    const int*   fut_ctxt = (const int*)d_in[0];
    const int*   fut_temp = (const int*)d_in[1];
    const float* enc      = (const float*)d_in[2];
    const float* emb_ctxt = (const float*)d_in[3];
    const float* emb_temp = (const float*)d_in[4];
    const float* attn_W   = (const float*)d_in[5];
    const float* attn_b   = (const float*)d_in[6];
    const float* v        = (const float*)d_in[7];
    const float* w_ih     = (const float*)d_in[8];
    // d_in[9] = w_hh (unused: h0 == 0)
    const float* b_ih     = (const float*)d_in[10];
    const float* b_hh     = (const float*)d_in[11];
    const float* out_W    = (const float*)d_in[12];
    const float* out_b    = (const float*)d_in[13];

    float* ws = (float*)d_ws;
    float*    q_proj   = ws;                              // 16384 floats
    float*    emb_t    = ws + 16384;                      // 16384
    uint32_t* we_frag  = (uint32_t*)(ws + 32768);         // 65536 u32
    float*    partials = ws + 32768 + 65536;              // 64*8*520
    float*    context  = partials + 64 * SPLIT * PART_STRIDE;  // 32768

    qproj_kernel<<<64, 256, 0, stream>>>(fut_ctxt, emb_ctxt, attn_W, attn_b, q_proj);
    embt_kernel<<<64, 256, 0, stream>>>(fut_temp, emb_temp, emb_t);
    wefrag_kernel<<<256, 32, 0, stream>>>(attn_W, we_frag);

    const size_t smem = (size_t)WE_U32 * 4 + 8 * PART_STRIDE * 4;  // ~278 KB (<320 KB WGP LDS)
    attn_kernel<<<dim3(SPLIT, 64), 256, smem, stream>>>(enc, v, q_proj, we_frag, partials);

    combine_kernel<<<64, 256, 0, stream>>>(partials, context);
    gru_kernel<<<64, 256, 0, stream>>>(emb_t, context, w_ih, b_ih, b_hh, (float*)d_out);
    head_kernel<<<64, 32, 0, stream>>>(out_W, out_b, (float*)d_out);
}
